// BatchRNN_24094766531256
// MI455X (gfx1250) — matrix operand order repaired and checked by reading, not compiled, not run on hardware
//
#include <hip/hip_runtime.h>

#define T_   1024
#define B_   64
#define H_   512
#define HID_ 512
#define G4_  2048
#define TB_  (T_ * B_)
#define EPS_ 1e-5f

typedef __attribute__((ext_vector_type(16))) __bf16 v16bf;
typedef __attribute__((ext_vector_type(8)))  float  v8f;

// ---- workspace layout (bytes) ----
// 0        : sumx  [512] f32
// 2048     : sumx2 [512] f32
// 4096     : a2    [512] f32
// 6144     : b2v   [512] f32
// 8192     : sync  u32 (256B slot)
// 65536    : y_bf  [T][B][H]       bf16 (64 MB)
// +67108864: wc    [2][2048][1024] bf16 ( 8 MB)  fused [Wx;Wh], n-major
// +8388608 : h_bf  [2][B][HID]     bf16 (128 KB)
#define WS_YOFF  65536UL
#define WS_WCOFF (WS_YOFF + (size_t)TB_ * H_ * 2)
#define WS_HOFF  (WS_WCOFF + (size_t)2 * G4_ * 1024 * 2)

// LDS B-slice: 64 rows (4 gates x 16 units), stride padded to 1032 bf16
#define BSTRIDE 1032
#define BSH_BYTES (64 * BSTRIDE * 2)   // 132096

__device__ __forceinline__ unsigned short f2bf(float x) {
  unsigned u = __builtin_bit_cast(unsigned, x);
  unsigned r = (u + 0x7FFFu + ((u >> 16) & 1u)) >> 16;  // RNE
  return (unsigned short)r;
}
__device__ __forceinline__ float sigm(float x) { return 1.0f / (1.0f + __expf(-x)); }
__device__ __forceinline__ float tanh_fast(float x) {
  // exact: tanh(x) = 1 - 2/(e^{2x}+1); saturates correctly at +/-inf
  float e = __expf(2.0f * x);
  return 1.0f - 2.0f / (e + 1.0f);
}

// ---------- init: zero stats/sync + h0 ----------
__global__ void k_init(unsigned* wsu, unsigned* hbfu) {
  int idx = blockIdx.x * blockDim.x + threadIdx.x;
  if (idx < 2112) wsu[idx] = 0u;                         // bytes [0, 8448)
  if (idx < 32768) hbfu[idx] = 0u;                       // h_bf = 0
}

__global__ void k_zero_out(float* out, long n) {
  long i = (long)blockIdx.x * blockDim.x + threadIdx.x;
  long stride = (long)gridDim.x * blockDim.x;
  for (; i < n; i += stride) out[i] = 0.0f;
}

// ---------- batch-norm statistics ----------
__global__ void k_stats(const float* __restrict__ x, float* sumx, float* sumx2) {
  int t = threadIdx.x;            // 256 threads, 2 fixed columns each -> coalesced
  int c0 = t, c1 = t + 256;
  size_t r0 = (size_t)blockIdx.x * 256;
  float s0 = 0.f, s1 = 0.f, q0 = 0.f, q1 = 0.f;
  for (int r = 0; r < 256; ++r) {
    const float* row = x + (r0 + r) * H_;
    float a = row[c0], b = row[c1];
    s0 += a; s1 += b; q0 += a * a; q1 += b * b;
  }
  atomicAdd(&sumx[c0], s0);  atomicAdd(&sumx[c1], s1);
  atomicAdd(&sumx2[c0], q0); atomicAdd(&sumx2[c1], q1);
}

__global__ void k_finalize(const float* sumx, const float* sumx2,
                           const float* scale, const float* bias,
                           const int* lengths, const int* training,
                           float* a2, float* b2v) {
  int h = threadIdx.x;  // 512
  float nz = 0.f;
  for (int b = 0; b < B_; ++b) nz += (float)lengths[b];
  float mean, var;
  if (*training) {
    mean = sumx[h] / nz;
    float m2 = sumx2[h] / nz;
    var = fmaxf(0.0f, m2 - mean * mean);
  } else { mean = 0.0f; var = 1.0f; }
  float a = scale[h] * rsqrtf(var + EPS_);
  a2[h]  = a;
  b2v[h] = bias[h] - mean * a;
}

// ---------- normalize + mask + f32->bf16 ----------
__global__ void k_norm(const float* __restrict__ x, const int* __restrict__ lengths,
                       const float* __restrict__ a2, const float* __restrict__ b2v,
                       unsigned short* __restrict__ ybf) {
  size_t i = ((size_t)blockIdx.x * blockDim.x + threadIdx.x) * 2;
  int hcol = (int)(i & (H_ - 1));
  size_t row = i >> 9;            // / H_
  int t = (int)(row >> 6);        // / B_
  int b = (int)(row & (B_ - 1));
  float m = (t < lengths[b]) ? 1.0f : 0.0f;
  float v0 = (x[i]     * a2[hcol]     + b2v[hcol])     * m;
  float v1 = (x[i + 1] * a2[hcol + 1] + b2v[hcol + 1]) * m;
  unsigned pack = (unsigned)f2bf(v0) | ((unsigned)f2bf(v1) << 16);
  *(unsigned*)(ybf + i) = pack;
}

// ---------- fused weight transpose [Wx;Wh] -> bf16 [d][n][k] ----------
__global__ void k_weights(const float* __restrict__ Wx_f, const float* __restrict__ Wh_f,
                          const float* __restrict__ Wx_b, const float* __restrict__ Wh_b,
                          unsigned short* __restrict__ wc) {
  size_t i = (size_t)blockIdx.x * blockDim.x + threadIdx.x;  // 2*2048*1024
  int k = (int)(i & 1023);
  int n = (int)((i >> 10) & (G4_ - 1));
  int d = (int)(i >> 21);
  const float* Wx = d ? Wx_b : Wx_f;
  const float* Wh = d ? Wh_b : Wh_f;
  float v = (k < H_) ? Wx[(size_t)k * G4_ + n] : Wh[(size_t)(k - H_) * G4_ + n];
  wc[i] = f2bf(v);
}

// ---------- persistent bidirectional LSTM ----------
// grid = 64 blocks (2 dirs x 32 j-tiles), 128 threads (4 waves = 4 m-tiles)
// B-slice (4 gates x 16 units x K=1024) lives in LDS for all 1024 steps.
extern __shared__ unsigned short bsh[];

__launch_bounds__(128, 1)
__global__ void k_rnn(const unsigned short* __restrict__ ybf,
                      const unsigned short* __restrict__ wc,
                      unsigned short* __restrict__ hbf,
                      const float* __restrict__ bfw, const float* __restrict__ bbw,
                      const int* __restrict__ lengths,
                      float* __restrict__ out,
                      unsigned int* __restrict__ syncc) {
  const int lane = threadIdx.x & 31;
  const int mt   = threadIdx.x >> 5;       // m-tile (batch/16)
  const int d    = blockIdx.x >> 5;        // direction
  const int jt   = blockIdx.x & 31;        // hidden-unit tile
  const int laneRow = lane & 15;
  const int half    = lane >> 4;

  // ---- stage block's weight slice into LDS (once; reused 1024 steps) ----
  // 64 rows (g*16+l) x 1024 bf16, LDS row stride BSTRIDE (bank-quad padded)
  for (int idx = threadIdx.x; idx < 64 * 128; idx += 128) {
    int row = idx >> 7;                    // 0..63 : g = row>>4, l = row&15
    int off = (idx & 127) * 8;             // 16B chunks
    const unsigned short* src =
        wc + (size_t)(d * G4_ + (row >> 4) * HID_ + jt * 16 + (row & 15)) * 1024 + off;
    *(uint4*)(&bsh[row * BSTRIDE + off]) = *(const uint4*)src;
  }
  __syncthreads();

  const int bmA  = mt * 16 + laneRow;      // A-fragment batch row
  const int lenA = lengths[bmA];
  const int nCol = jt * 16 + laneRow;      // B row index == C column j
  const float* bias = d ? bbw : bfw;
  const float bias_g0 = bias[0 * HID_ + nCol];
  const float bias_g1 = bias[1 * HID_ + nCol];
  const float bias_g2 = bias[2 * HID_ + nCol];
  const float bias_g3 = bias[3 * HID_ + nCol];

  int bmC[8], lenC[8];
#pragma unroll
  for (int r = 0; r < 8; ++r) { bmC[r] = mt * 16 + 8 * half + r; lenC[r] = lengths[bmC[r]]; }

  // two loop-invariant LDS bases so every ds_load gets an immediate offset:
  //   pB0 -> gate0 (gate1 at +16*BSTRIDE elems = 33024B, fits 16-bit imm)
  //   pB2 -> gate2 (gate3 at +16*BSTRIDE elems)
  const unsigned short* pB0 = &bsh[(0 * 16 + laneRow) * BSTRIDE + half * 16];
  const unsigned short* pB2 = &bsh[(2 * 16 + laneRow) * BSTRIDE + half * 16];
#define GOFF (16 * BSTRIDE)

  const unsigned short* hrowA = hbf + (size_t)(d * B_ + bmA) * HID_;
  unsigned short* hwr = hbf + (size_t)d * B_ * HID_;

  v8f cst;
#pragma unroll
  for (int r = 0; r < 8; ++r) cst[r] = 0.0f;

  for (int s = 0; s < T_; ++s) {
    const int tA = d ? ((lenA - 1 - s) & (T_ - 1)) : s;   // length-aware flip, fused
    const unsigned short* xrow = ybf + ((size_t)tA * B_ + bmA) * H_;
    if (s + 1 < T_) {                                     // prefetch next x row
      int tN = d ? ((lenA - 2 - s) & (T_ - 1)) : s + 1;
      __builtin_prefetch(ybf + ((size_t)tN * B_ + bmA) * H_ + half * 8, 0, 1);
    }

    v8f acc0, acc1, acc2, acc3;
#pragma unroll
    for (int r = 0; r < 8; ++r) { acc0[r] = bias_g0; acc1[r] = bias_g1; acc2[r] = bias_g2; acc3[r] = bias_g3; }

    // z = [x_t ; h] @ [Wx ; Wh] + b ; K=1024 in two 512-halves (x then h)
#pragma unroll 1
    for (int seg = 0; seg < 2; ++seg) {
      const unsigned short* asrc = seg ? hrowA : xrow;
      const int ksegbase = seg * 512;
#pragma unroll 4
      for (int kk = 0; kk < 16; ++kk) {
        const int ak = kk * 32;                 // within segment
        const int kb = ksegbase + ak;           // within fused K (LDS B index)
        union { v16bf v; uint4 q[2]; } A, Bq0, Bq1, Bq2, Bq3;
        // A 16x32 bf16: chunks at k+half*16 and k+half*16+8
        A.q[0] = *(const uint4*)(asrc + ak + half * 16);
        A.q[1] = *(const uint4*)(asrc + ak + half * 16 + 8);
        // B 32x16 bf16 from LDS: 32 contiguous bytes per lane, imm offsets only
        Bq0.q[0] = *(const uint4*)(pB0 + kb);        Bq0.q[1] = *(const uint4*)(pB0 + kb + 8);
        Bq1.q[0] = *(const uint4*)(pB0 + GOFF + kb); Bq1.q[1] = *(const uint4*)(pB0 + GOFF + kb + 8);
        Bq2.q[0] = *(const uint4*)(pB2 + kb);        Bq2.q[1] = *(const uint4*)(pB2 + kb + 8);
        Bq3.q[0] = *(const uint4*)(pB2 + GOFF + kb); Bq3.q[1] = *(const uint4*)(pB2 + GOFF + kb + 8);
        acc0 = __builtin_amdgcn_wmma_f32_16x16x32_bf16(false, A.v, false, Bq0.v, (short)0, acc0, false, false);
        acc1 = __builtin_amdgcn_wmma_f32_16x16x32_bf16(false, A.v, false, Bq1.v, (short)0, acc1, false, false);
        acc2 = __builtin_amdgcn_wmma_f32_16x16x32_bf16(false, A.v, false, Bq2.v, (short)0, acc2, false, false);
        acc3 = __builtin_amdgcn_wmma_f32_16x16x32_bf16(false, A.v, false, Bq3.v, (short)0, acc3, false, false);
      }
    }

    // gate math fully in registers; c persists in VGPRs across all steps
#pragma unroll
    for (int r = 0; r < 8; ++r) {
      float cn = sigm(acc1[r]) * cst[r] + sigm(acc0[r]) * tanh_fast(acc2[r]);
      float hn = sigm(acc3[r]) * tanh_fast(cn);
      cst[r] = cn;
      hwr[(size_t)bmC[r] * HID_ + nCol] = f2bf(hn);                       // recurrent state
      int tOut = d ? ((lenC[r] - 1 - s) & (T_ - 1)) : s;                  // fused output flip
      atomicAdd(out + ((size_t)tOut * B_ + bmC[r]) * HID_ + nCol, hn);    // fwd+bwd merge
    }

    // one device-wide sync per timestep (monotonic counter, reset each launch)
    __threadfence();
    __syncthreads();
    if (threadIdx.x == 0) {
      atomicAdd(syncc, 1u);
      unsigned target = 64u * (unsigned)(s + 1);
      while (atomicAdd(syncc, 0u) < target) __builtin_amdgcn_s_sleep(1);
    }
    __syncthreads();
    __threadfence();
  }
}

extern "C" void kernel_launch(void* const* d_in, const int* in_sizes, int n_in,
                              void* d_out, int out_size, void* d_ws, size_t ws_size,
                              hipStream_t stream) {
  const float* x        = (const float*)d_in[0];
  const int*   lengths  = (const int*)d_in[1];
  const float* scale    = (const float*)d_in[3];
  const float* bias     = (const float*)d_in[4];
  const float* Wx_f     = (const float*)d_in[5];
  const float* Wh_f     = (const float*)d_in[6];
  const float* b_f      = (const float*)d_in[7];
  const float* Wx_b     = (const float*)d_in[8];
  const float* Wh_b     = (const float*)d_in[9];
  const float* b_b      = (const float*)d_in[10];
  const int*   training = (const int*)d_in[11];
  float* out = (float*)d_out;

  char* ws = (char*)d_ws;
  float*          sumx  = (float*)(ws + 0);
  float*          sumx2 = (float*)(ws + 2048);
  float*          a2    = (float*)(ws + 4096);
  float*          b2v   = (float*)(ws + 6144);
  unsigned int*   syncc = (unsigned int*)(ws + 8192);
  unsigned short* ybf   = (unsigned short*)(ws + WS_YOFF);
  unsigned short* wc    = (unsigned short*)(ws + WS_WCOFF);
  unsigned short* hbf   = (unsigned short*)(ws + WS_HOFF);

  k_init<<<128, 256, 0, stream>>>((unsigned*)ws, (unsigned*)hbf);
  k_zero_out<<<1024, 256, 0, stream>>>(out, (long)out_size);
  k_stats<<<256, 256, 0, stream>>>(x, sumx, sumx2);
  k_finalize<<<1, 512, 0, stream>>>(sumx, sumx2, scale, bias, lengths, training, a2, b2v);
  k_norm<<<65536, 256, 0, stream>>>(x, lengths, a2, b2v, ybf);
  k_weights<<<16384, 256, 0, stream>>>(Wx_f, Wh_f, Wx_b, Wh_b, wc);
  k_rnn<<<64, 128, BSH_BYTES, stream>>>(ybf, wc, hbf, b_f, b_b, lengths, out, syncc);
}
